// _SoftTreeModule_68478958567426
// MI455X (gfx1250) — matrix operand order repaired and checked by reading, not compiled, not run on hardware
//
#include <hip/hip_runtime.h>
#include <hip/hip_bf16.h>
#include <hip/hip_fp16.h>

typedef __attribute__((ext_vector_type(16))) _Float16 v16h;
typedef __attribute__((ext_vector_type(8)))  _Float16 v8h;
typedef __attribute__((ext_vector_type(8)))  float    v8f;

#define DEPTH   8
#define NFEAT   512
#define NCLASS  1000
#define NLEAF   256
#define NINT    255
#define BATCH   16384
#define MTILE   128
#define NPAD    1024   // classes padded to multiple of 128

// LDS carve-out (dynamic):
//   p-tile  f32 [128][256] = 128KB  (aliased by 2x 64KB leafT chunk buffers in stage 3)
//   mu      f16 [128][256] =  64KB
//   w chunk f16 2x[256][32] = 32KB  (double buffered)
#define LDS_P_OFF    0
#define LDS_MU_OFF   (128 * 256 * 4)
#define LDS_W_OFF    (LDS_MU_OFF + 128 * 256 * 2)
#define LDS_TOTAL    (LDS_W_OFF + 2 * 256 * 32 * 2)

// Workspace layout: leafT f16 [NPAD][NLEAF] then gate_wh f16 [256][512]
#define WS_LEAFT_OFF 0
#define WS_GWH_OFF   ((size_t)NPAD * NLEAF * 2)

// ---------------------------------------------------------------------------
// CDNA5 async global->LDS copy (ASYNCcnt) with synchronous fallback.
// Builtin signature (from hipcc diagnostic): param 1 is
//   'int __attribute__((vector_size(16))) __device__ *'  (global AS, int4)
// ---------------------------------------------------------------------------
#if defined(__has_builtin)
#if __has_builtin(__builtin_amdgcn_global_load_async_to_lds_b128) && \
    __has_builtin(__builtin_amdgcn_s_wait_asynccnt)
#define HAVE_ASYNC_LDS 1
#endif
#endif
#ifndef HAVE_ASYNC_LDS
#define HAVE_ASYNC_LDS 0
#endif

typedef __attribute__((__vector_size__(4 * sizeof(int)))) int i4x;

__device__ __forceinline__ void async_copy_b128(const _Float16* gsrc, _Float16* ldst) {
#if HAVE_ASYNC_LDS
    __builtin_amdgcn_global_load_async_to_lds_b128(
        (__attribute__((address_space(1))) i4x*)(const void*)gsrc,
        (__attribute__((address_space(3))) i4x*)(void*)ldst,
        0, 0);
#else
    *(uint4*)ldst = *(const uint4*)gsrc;
#endif
}

__device__ __forceinline__ void wait_async_copies() {
#if HAVE_ASYNC_LDS
    __builtin_amdgcn_s_wait_asynccnt(0);
#endif
}

// ---------------------------------------------------------------------------
// Prep 1: leafT[n][l] = softmax(leaf_logits[l])[n], f16, transposed, n padded.
// ---------------------------------------------------------------------------
__global__ void leaf_softmax_kernel(const float* __restrict__ leaf_logits,
                                    _Float16* __restrict__ leafT) {
    __shared__ float red[256];
    const int l = blockIdx.x;          // leaf index 0..255
    const int t = threadIdx.x;         // 0..255
    const float* row = leaf_logits + (size_t)l * NCLASS;

    float mx = -3.4e38f;
    for (int n = t; n < NCLASS; n += 256) mx = fmaxf(mx, row[n]);
    red[t] = mx; __syncthreads();
    for (int s = 128; s > 0; s >>= 1) { if (t < s) red[t] = fmaxf(red[t], red[t + s]); __syncthreads(); }
    mx = red[0]; __syncthreads();

    float sum = 0.f;
    for (int n = t; n < NCLASS; n += 256) sum += __expf(row[n] - mx);
    red[t] = sum; __syncthreads();
    for (int s = 128; s > 0; s >>= 1) { if (t < s) red[t] += red[t + s]; __syncthreads(); }
    const float inv = 1.0f / red[0];

    for (int n = t; n < NPAD; n += 256) {
        float v = (n < NCLASS) ? __expf(row[n] - mx) * inv : 0.f;
        leafT[(size_t)n * NLEAF + l] = (_Float16)v;
    }
}

// ---------------------------------------------------------------------------
// Prep 2: gate_w (f32, 255x512) -> gate_wh (f16, 256x512, row 255 zero-padded)
// ---------------------------------------------------------------------------
__global__ void gate_w_cvt_kernel(const float* __restrict__ gate_w,
                                  _Float16* __restrict__ gate_wh) {
    const int i = blockIdx.x * 256 + threadIdx.x;   // 0 .. 256*512-1
    const int n = i >> 9;
    _Float16 v;
    if (n < NINT) v = (_Float16)gate_w[i];
    else          v = (_Float16)0.f;
    gate_wh[i] = v;
}

// ---------------------------------------------------------------------------
// Fused soft-decision-tree kernel: one 128-row batch tile per workgroup.
// ---------------------------------------------------------------------------
__global__ __launch_bounds__(256) void soft_tree_fused(
        const float* __restrict__ x,
        const _Float16* __restrict__ gate_wh,
        const float* __restrict__ gate_b,
        const _Float16* __restrict__ leafT,
        float* __restrict__ out) {
    extern __shared__ __align__(128) char smem[];
    float*    lds_p    = (float*)(smem + LDS_P_OFF);
    _Float16* lds_mu   = (_Float16*)(smem + LDS_MU_OFF);
    _Float16* lds_w    = (_Float16*)(smem + LDS_W_OFF);       // 2 buffers of 256*32
    _Float16* lds_leaf = (_Float16*)(smem + LDS_P_OFF);       // 2 buffers of 128*256

    const int t       = threadIdx.x;
    const int lane    = t & 31;
    const int wave    = t >> 5;
    const int lrow    = lane & 15;   // row (A/C) or column (B) within 16x16 tile
    const int lhi     = lane >> 4;   // K-half selector per WMMA operand layout
    const int m_strip = wave * 16;   // this wave's M strip inside the tile
    const int m_base  = blockIdx.x * MTILE;

    // ============ Stage 1: logits = x @ gate_w^T, p = sigmoid(logits + b) ====
    v8f acc[16] = {};

    // copy gate_wh K-chunk kc -> wbuf (pure f16 copy, 64B per thread, async)
    auto copy_w_chunk = [&](int kc, _Float16* wdst) {
        const _Float16* src = gate_wh + (size_t)t * NFEAT + kc;
        _Float16* dst = wdst + t * 32;
        #pragma unroll
        for (int q = 0; q < 4; ++q) async_copy_b128(src + q * 8, dst + q * 8);
    };

    copy_w_chunk(0, lds_w);
    wait_async_copies();
    __syncthreads();

    for (int i = 0; i < NFEAT / 32; ++i) {
        const int kc = i * 32;
        const _Float16* wcur = lds_w + (i & 1) * (256 * 32);

        // overlap: async-stage next w chunk while computing on current
        if (i + 1 < NFEAT / 32)
            copy_w_chunk(kc + 32, lds_w + ((i + 1) & 1) * (256 * 32));

        // A fragment straight from global x with f32->f16 convert.
        // Layout: lanes 0-15 hold K = kc+{0..15}, lanes 16-31 K = kc+{16..31}.
        v16h afrag;
        {
            const float* xr = x + (size_t)(m_base + m_strip + lrow) * NFEAT + kc + lhi * 16;
            if (kc + 32 < NFEAT) __builtin_prefetch(xr + 32, 0, 1);
            float4 a0 = *(const float4*)(xr + 0);
            float4 a1 = *(const float4*)(xr + 4);
            float4 a2 = *(const float4*)(xr + 8);
            float4 a3 = *(const float4*)(xr + 12);
            afrag[0]  = (_Float16)a0.x; afrag[1]  = (_Float16)a0.y;
            afrag[2]  = (_Float16)a0.z; afrag[3]  = (_Float16)a0.w;
            afrag[4]  = (_Float16)a1.x; afrag[5]  = (_Float16)a1.y;
            afrag[6]  = (_Float16)a1.z; afrag[7]  = (_Float16)a1.w;
            afrag[8]  = (_Float16)a2.x; afrag[9]  = (_Float16)a2.y;
            afrag[10] = (_Float16)a2.z; afrag[11] = (_Float16)a2.w;
            afrag[12] = (_Float16)a3.x; afrag[13] = (_Float16)a3.y;
            afrag[14] = (_Float16)a3.z; afrag[15] = (_Float16)a3.w;
        }

        // B fragments double-buffered so ds_loads of tile nt+1 overlap wmma nt
        v16h bf = *(const v16h*)(wcur + (0 * 16 + lrow) * 32 + lhi * 16);
        #pragma unroll
        for (int nt = 0; nt < 16; ++nt) {
            const v16h bn = (nt < 15)
                ? *(const v16h*)(wcur + ((nt + 1) * 16 + lrow) * 32 + lhi * 16)
                : bf;
            acc[nt] = __builtin_amdgcn_wmma_f32_16x16x32_f16(
                false, afrag, false, bf, (short)0, acc[nt], false, false);
            bf = bn;
        }

        wait_async_copies();
        __syncthreads();
    }

    // bias + sigmoid -> p-tile in LDS
    #pragma unroll
    for (int nt = 0; nt < 16; ++nt) {
        const int n = nt * 16 + lrow;                 // node index (C-layout: lane = N)
        const float bias = (n < NINT) ? gate_b[n] : 0.f;
        #pragma unroll
        for (int r = 0; r < 8; ++r) {
            const float v = acc[nt][r] + bias;
            const float p = 1.0f / (1.0f + __expf(-v));
            const int m = m_strip + lhi * 8 + r;      // C-layout: M = r + 8*(lane>=16)
            lds_p[m * 256 + n] = p;
        }
    }
    __syncthreads();

    // ============ Stage 2: tree product -> mu (f16 in LDS) ===================
    for (int idx = t; idx < MTILE * NLEAF; idx += 256) {
        const int b = idx >> 8;
        const int l = idx & 255;
        float mu = 1.0f;
        #pragma unroll
        for (int d = 0; d < DEPTH; ++d) {
            const int node = (1 << d) - 1 + (l >> (DEPTH - d));
            const int bit  = (l >> (DEPTH - 1 - d)) & 1;
            const float g  = lds_p[b * 256 + node];
            mu *= bit ? g : (1.0f - g);
        }
        lds_mu[idx] = (_Float16)mu;
    }
    __syncthreads();  // p-tile dead; leaf buffers may now overwrite it

    // ============ Stage 3: out = mu @ leaf_dists (N in chunks of 128) ========
    // copy leafT rows n0..n0+127 (full K=256) -> one of the two 64KB buffers
    auto copy_leaf_chunk = [&](int n0, _Float16* ldst) {
        const int rrow = t >> 1;
        const int part = (t & 1) * 128;
        const _Float16* src = leafT + (size_t)(n0 + rrow) * NLEAF + part;
        _Float16* dst = ldst + rrow * NLEAF + part;
        #pragma unroll
        for (int q = 0; q < 16; ++q) async_copy_b128(src + q * 8, dst + q * 8);
    };

    copy_leaf_chunk(0, lds_leaf);
    wait_async_copies();
    __syncthreads();

    for (int nc = 0; nc < NPAD / 128; ++nc) {
        const int n0 = nc * 128;
        const _Float16* lcur = lds_leaf + (nc & 1) * (128 * 256);

        // overlap: async-stage next leafT chunk while computing on current
        if (nc + 1 < NPAD / 128)
            copy_leaf_chunk(n0 + 128, lds_leaf + ((nc + 1) & 1) * (128 * 256));

        v8f acc2[8] = {};
        #pragma unroll
        for (int kc = 0; kc < NLEAF; kc += 32) {
            v16h afrag;
            {
                const _Float16* mr = lds_mu + (m_strip + lrow) * 256 + kc + lhi * 16;
                v8h lo = *(const v8h*)(mr);
                v8h hi = *(const v8h*)(mr + 8);
                #pragma unroll
                for (int j = 0; j < 8; ++j) { afrag[j] = lo[j]; afrag[j + 8] = hi[j]; }
            }
            v16h bf = *(const v16h*)(lcur + (0 * 16 + lrow) * 256 + kc + lhi * 16);
            #pragma unroll
            for (int nt = 0; nt < 8; ++nt) {
                const v16h bn = (nt < 7)
                    ? *(const v16h*)(lcur + ((nt + 1) * 16 + lrow) * 256 + kc + lhi * 16)
                    : bf;
                acc2[nt] = __builtin_amdgcn_wmma_f32_16x16x32_f16(
                    false, afrag, false, bf, (short)0, acc2[nt], false, false);
                bf = bn;
            }
        }

        #pragma unroll
        for (int nt = 0; nt < 8; ++nt) {
            const int col = n0 + nt * 16 + lrow;
            if (col < NCLASS) {
                #pragma unroll
                for (int r = 0; r < 8; ++r) {
                    const int m = m_base + m_strip + lhi * 8 + r;
                    out[(size_t)m * NCLASS + col] = acc2[nt][r];
                }
            }
        }

        wait_async_copies();
        __syncthreads();  // next chunk's buffer is now safe to consume
    }
}

// ---------------------------------------------------------------------------
extern "C" void kernel_launch(void* const* d_in, const int* in_sizes, int n_in,
                              void* d_out, int out_size, void* d_ws, size_t ws_size,
                              hipStream_t stream) {
    const float* x           = (const float*)d_in[0];
    const float* gate_w      = (const float*)d_in[1];
    const float* gate_b      = (const float*)d_in[2];
    const float* leaf_logits = (const float*)d_in[3];
    float* out = (float*)d_out;

    _Float16* leafT   = (_Float16*)((char*)d_ws + WS_LEAFT_OFF); // 512 KB
    _Float16* gate_wh = (_Float16*)((char*)d_ws + WS_GWH_OFF);   // 256 KB

    (void)in_sizes; (void)n_in; (void)out_size; (void)ws_size;

    leaf_softmax_kernel<<<NLEAF, 256, 0, stream>>>(leaf_logits, leafT);
    gate_w_cvt_kernel<<<(256 * NFEAT) / 256, 256, 0, stream>>>(gate_w, gate_wh);

    (void)hipFuncSetAttribute((const void*)soft_tree_fused,
                              hipFuncAttributeMaxDynamicSharedMemorySize, LDS_TOTAL);
    soft_tree_fused<<<BATCH / MTILE, 256, LDS_TOTAL, stream>>>(
        x, gate_wh, gate_b, leafT, out);
}
